// GCN_88742614270553
// MI455X (gfx1250) — matrix operand order repaired and checked by reading, not compiled, not run on hardware
//
#include <hip/hip_runtime.h>
#include <hip/hip_bf16.h>

typedef __attribute__((ext_vector_type(16))) __bf16 v16bf;
typedef __attribute__((ext_vector_type(8)))  float  v8f;
typedef __attribute__((ext_vector_type(4)))  int    v4i;

#define NN   50000
#define NE   800000
#define IND  256
#define HID  128
#define OUTD 64
#define NBR  7

// gfx1250 async global->LDS DMA (ASYNCcnt-tracked), guarded per-toolchain.
// Probe-verified param types: (v4i addrspace(1)*, v4i addrspace(3)*, imm, imm)
#if defined(__AMDGCN__) && __has_builtin(__builtin_amdgcn_global_load_async_to_lds_b128) && \
    __has_builtin(__builtin_amdgcn_s_wait_asynccnt)
#define HAVE_ASYNC_LDS 1
typedef v4i __attribute__((address_space(1)))* gv4i_p;
typedef v4i __attribute__((address_space(3)))* lv4i_p;
#else
#define HAVE_ASYNC_LDS 0
#endif

__device__ __forceinline__ v8f wmma_bf16(v16bf a, v16bf b, v8f c) {
  // v_wmma_f32_16x16x32_bf16: (neg_a, A, neg_b, B, c_mod, C, reuse_a, reuse_b)
  return __builtin_amdgcn_wmma_f32_16x16x32_bf16(false, a, false, b, (short)0, c,
                                                 false, false);
}

// ---------------------------------------------------------------------------
// Pack weights (optionally summed over branches) into per-lane WMMA B layout,
// split into bf16 hi + lo residual.  Dest idx = ((kc*NT + nt)*32 + lane)*16 + j
// where element = W[kc*32 + (lane>>4)*16 + j][nt*16 + (lane&15)].
// ---------------------------------------------------------------------------
template <int K_DIM, int N_OUT>
__global__ __launch_bounds__(256) void pack_weights(const float* __restrict__ W,
                                                    int n_branch, int mat_stride,
                                                    __bf16* __restrict__ Whi,
                                                    __bf16* __restrict__ Wlo) {
  constexpr int NT  = N_OUT / 16;
  constexpr int TOT = (K_DIM / 32) * NT * 32 * 16;
  int idx = blockIdx.x * 256 + threadIdx.x;
  if (idx >= TOT) return;
  int j    = idx & 15;
  int lane = (idx >> 4) & 31;
  int rem  = idx >> 9;
  int nt   = rem % NT;
  int kc   = rem / NT;
  int k    = kc * 32 + (lane >> 4) * 16 + j;
  int n    = nt * 16 + (lane & 15);
  float s = 0.0f;
  for (int b = 0; b < n_branch; ++b) s += W[b * mat_stride + k * N_OUT + n];
  __bf16 hi = (__bf16)s;
  Whi[idx] = hi;
  Wlo[idx] = (__bf16)(s - (float)hi);
}

__global__ void bias_sum_kernel(const float* __restrict__ b_br,
                                float* __restrict__ b_sum) {
  int i = threadIdx.x;
  if (i < HID) {
    float s = 0.0f;
    for (int b = 0; b < NBR; ++b) s += b_br[b * HID + i];
    b_sum[i] = s;
  }
}

// out[i] = bias[i & (D-1)]  (row-broadcast init of the scatter accumulator)
template <int D>
__global__ __launch_bounds__(256) void init_rows(float* __restrict__ out,
                                                 const float* __restrict__ bias,
                                                 int total) {
  static_assert((D & (D - 1)) == 0, "D must be pow2");
  int i = blockIdx.x * 256 + threadIdx.x;
  if (i < total) out[i] = bias[i & (D - 1)];
}

// ---------------------------------------------------------------------------
// GEMM: C[n_rows, N_OUT] = (RELU? relu(A) : A)[n_rows, K_DIM] @ B
// B pre-packed bf16 hi/lo.  fp32 recovered via hi*hi + hi*lo + lo*hi.
// 8 waves/block, one 16-row WMMA tile per wave, B chunk staged in LDS
// (async global->LDS DMA when available).
// ---------------------------------------------------------------------------
template <int K_DIM, int N_OUT, bool RELU>
__global__ __launch_bounds__(256) void gemm_wmma(const float* __restrict__ A,
                                                 const __bf16* __restrict__ Bhi,
                                                 const __bf16* __restrict__ Blo,
                                                 float* __restrict__ C, int n_rows) {
  constexpr int KC = K_DIM / 32;  // k-chunks of 32
  constexpr int NT = N_OUT / 16;  // 16-wide n-tiles
  __shared__ __align__(32) __bf16 sBhi[NT * 32 * 16];
  __shared__ __align__(32) __bf16 sBlo[NT * 32 * 16];

  const int tid  = threadIdx.x;
  const int wave = tid >> 5;
  const int lane = tid & 31;
  const int rowBase = blockIdx.x * 128 + wave * 16;

  // A-operand addressing (contiguous k order, same as packed B):
  //  lane<16 holds K [0..15]; lane>=16 holds K [16..31]
  int arow = rowBase + (lane & 15);
  if (arow >= n_rows) arow = n_rows - 1;  // clamp; stores are guarded below
  const float* arowp = A + (size_t)arow * K_DIM;
  const int khalf = (lane >> 4) * 16;

  v8f acc[NT];
  const v8f vzero = {0.f, 0.f, 0.f, 0.f, 0.f, 0.f, 0.f, 0.f};
#pragma unroll
  for (int nt = 0; nt < NT; ++nt) acc[nt] = vzero;

  for (int kc = 0; kc < KC; ++kc) {
    // --- stage packed B chunk (hi+lo) into LDS ---
    __syncthreads();  // protect previous iteration's LDS reads
    {
      const uint4* ghi = reinterpret_cast<const uint4*>(Bhi) + kc * (NT * 64);
      const uint4* glo = reinterpret_cast<const uint4*>(Blo) + kc * (NT * 64);
      uint4* shi = reinterpret_cast<uint4*>(sBhi);
      uint4* slo = reinterpret_cast<uint4*>(sBlo);
#if HAVE_ASYNC_LDS
#pragma unroll
      for (int u = tid; u < NT * 64; u += 256) {
        __builtin_amdgcn_global_load_async_to_lds_b128((gv4i_p)(ghi + u),
                                                       (lv4i_p)(shi + u), 0, 0);
        __builtin_amdgcn_global_load_async_to_lds_b128((gv4i_p)(glo + u),
                                                       (lv4i_p)(slo + u), 0, 0);
      }
      __builtin_amdgcn_s_wait_asynccnt(0);  // own copies done before barrier
#else
      for (int u = tid; u < NT * 64; u += 256) {
        shi[u] = ghi[u];
        slo[u] = glo[u];
      }
#endif
    }
    __syncthreads();

    if (kc + 1 < KC)  // global_prefetch_b8 for next A chunk
      __builtin_prefetch(arowp + (kc + 1) * 32 + khalf, 0, 1);

    // --- build A operand (two contiguous groups of 8 f32 per lane) ---
    float t[16];
    {
      const float4* p0 = reinterpret_cast<const float4*>(arowp + kc * 32 + khalf);
      const float4* p1 = reinterpret_cast<const float4*>(arowp + kc * 32 + khalf + 8);
      float4 u0 = p0[0], u1 = p0[1], u2 = p1[0], u3 = p1[1];
      t[0] = u0.x; t[1] = u0.y; t[2]  = u0.z; t[3]  = u0.w;
      t[4] = u1.x; t[5] = u1.y; t[6]  = u1.z; t[7]  = u1.w;
      t[8] = u2.x; t[9] = u2.y; t[10] = u2.z; t[11] = u2.w;
      t[12] = u3.x; t[13] = u3.y; t[14] = u3.z; t[15] = u3.w;
    }
    v16bf ahi, alo;
#pragma unroll
    for (int j = 0; j < 16; ++j) {
      float v = t[j];
      if (RELU) v = fmaxf(v, 0.0f);
      __bf16 h = (__bf16)v;
      ahi[j] = h;
      alo[j] = (__bf16)(v - (float)h);
    }

    // --- WMMA over all n-tiles, B from LDS ---
#pragma unroll
    for (int nt = 0; nt < NT; ++nt) {
      const v16bf bhi = *reinterpret_cast<const v16bf*>(sBhi + (nt * 32 + lane) * 16);
      const v16bf blo = *reinterpret_cast<const v16bf*>(sBlo + (nt * 32 + lane) * 16);
      acc[nt] = wmma_bf16(ahi, bhi, acc[nt]);
      acc[nt] = wmma_bf16(ahi, blo, acc[nt]);
      acc[nt] = wmma_bf16(alo, bhi, acc[nt]);
    }
  }

  // --- store: C/D layout: VGPR r -> M = r + 8*(lane>>4), N = lane&15 ---
  const int mBase = rowBase + (lane >> 4) * 8;
  const int nCol  = lane & 15;
#pragma unroll
  for (int nt = 0; nt < NT; ++nt) {
#pragma unroll
    for (int r = 0; r < 8; ++r) {
      int row = mBase + r;
      if (row < n_rows) C[(size_t)row * N_OUT + nt * 16 + nCol] = acc[nt][r];
    }
  }
}

// ---------------------------------------------------------------------------
// Edge scatter: out[dst] += w * feat[src].  One wave per edge.
// feat/out are L2-resident (25.6 MB << 192 MB L2), atomics resolve in L2.
// ---------------------------------------------------------------------------
template <int D>
__global__ __launch_bounds__(256) void spmm_scatter(const int* __restrict__ src,
                                                    const int* __restrict__ dst,
                                                    const float* __restrict__ w,
                                                    const float* __restrict__ feat,
                                                    float* __restrict__ out,
                                                    int n_edges) {
  constexpr int VPL = D / 32;  // floats per lane
  int e = blockIdx.x * 8 + (threadIdx.x >> 5);
  if (e >= n_edges) return;
  int lane = threadIdx.x & 31;
  int s = src[e];
  int d = dst[e];
  float ww = w[e];
  const float* f = feat + (size_t)s * D + lane * VPL;
  float* o = out + (size_t)d * D + lane * VPL;
  if constexpr (VPL == 4) {
    float4 v = *reinterpret_cast<const float4*>(f);
    atomicAdd(&o[0], ww * v.x);
    atomicAdd(&o[1], ww * v.y);
    atomicAdd(&o[2], ww * v.z);
    atomicAdd(&o[3], ww * v.w);
  } else {
    float2 v = *reinterpret_cast<const float2*>(f);
    atomicAdd(&o[0], ww * v.x);
    atomicAdd(&o[1], ww * v.y);
  }
}

// ---------------------------------------------------------------------------
extern "C" void kernel_launch(void* const* d_in, const int* in_sizes, int n_in,
                              void* d_out, int out_size, void* d_ws, size_t ws_size,
                              hipStream_t stream) {
  const float* x     = (const float*)d_in[0];  // [NN, IND]
  const int*   esrc  = (const int*)d_in[1];    // [NE]
  const int*   edst  = (const int*)d_in[2];    // [NE]
  const float* ew    = (const float*)d_in[3];  // [NE]
  const float* W_br  = (const float*)d_in[4];  // [NBR, IND, HID]
  const float* b_br  = (const float*)d_in[5];  // [NBR, HID]
  const float* W8    = (const float*)d_in[6];  // [HID, OUTD]
  const float* b8    = (const float*)d_in[7];  // [OUTD]
  float* out = (float*)d_out;                  // [NN, OUTD]

  // workspace carve-up (256B aligned)
  char* ws = (char*)d_ws;
  size_t off = 0;
  auto carve = [&](size_t bytes) {
    char* p = ws + off;
    off = (off + bytes + 255) & ~(size_t)255;
    return p;
  };
  __bf16* W1hi   = (__bf16*)carve(IND * HID * sizeof(__bf16));
  __bf16* W1lo   = (__bf16*)carve(IND * HID * sizeof(__bf16));
  __bf16* W8hi   = (__bf16*)carve(HID * OUTD * sizeof(__bf16));
  __bf16* W8lo   = (__bf16*)carve(HID * OUTD * sizeof(__bf16));
  float* b_sum   = (float*)carve(HID * sizeof(float));
  float* support = (float*)carve((size_t)NN * HID * sizeof(float));
  float* acc     = (float*)carve((size_t)NN * HID * sizeof(float));
  float* pre2    = (float*)carve((size_t)NN * OUTD * sizeof(float));
  (void)ws_size;

  // 1) fold 7 branches into one weight/bias, pack for WMMA
  pack_weights<IND, HID><<<(IND * HID + 255) / 256, 256, 0, stream>>>(
      W_br, NBR, IND * HID, W1hi, W1lo);
  pack_weights<HID, OUTD><<<(HID * OUTD + 255) / 256, 256, 0, stream>>>(
      W8, 1, 0, W8hi, W8lo);
  bias_sum_kernel<<<1, 128, 0, stream>>>(b_br, b_sum);

  // 2) support = x @ W_sum
  gemm_wmma<IND, HID, false><<<(NN + 127) / 128, 256, 0, stream>>>(
      x, W1hi, W1lo, support, NN);

  // 3) acc = b_sum;  acc[dst] += w * support[src]
  init_rows<HID><<<((NN * HID) + 255) / 256, 256, 0, stream>>>(acc, b_sum, NN * HID);
  spmm_scatter<HID><<<(NE + 7) / 8, 256, 0, stream>>>(esrc, edst, ew, support, acc, NE);

  // 4) pre2 = relu(acc) @ W8
  gemm_wmma<HID, OUTD, true><<<(NN + 127) / 128, 256, 0, stream>>>(
      acc, W8hi, W8lo, pre2, NN);

  // 5) out = b8;  out[dst] += w * pre2[src]
  init_rows<OUTD><<<((NN * OUTD) + 255) / 256, 256, 0, stream>>>(out, b8, NN * OUTD);
  spmm_scatter<OUTD><<<(NE + 7) / 8, 256, 0, stream>>>(esrc, edst, ew, pre2, out, NE);
}
